// Attention_26620207301186
// MI455X (gfx1250) — matrix operand order repaired and checked by reading, not compiled, not run on hardware
//
#include <hip/hip_runtime.h>
#include <hip/hip_bf16.h>

// ---------------------------------------------------------------------------
// Attention (B=4, S=2048, D=1024) for gfx1250 (MI455X, wave32, WMMA).
// Pipeline: 3x bf16 projection GEMMs -> streaming softmax (online max/sum,
// QK^T recomputed in pass 2) -> bf16 P*V GEMM with fp32 output.
// All matrix math via v_wmma_f32_16x16x32_bf16 (fp32 accumulate).
// B tiles staged n-major in LDS so A and B fragment loads are contiguous
// 16B runs per lane (ds_load_b128). Global->register prefetch pipelines the
// next k-tile behind the current tile's WMMAs; global_prefetch_b8 streams
// the HBM-resident activations three tiles ahead.
// ---------------------------------------------------------------------------

typedef __attribute__((ext_vector_type(16))) __bf16 v16bf;
typedef __attribute__((ext_vector_type(8)))  float  v8f;

union FragBF   { v16bf v; __bf16 e[16]; };
union PackBF16 { uint4 u[2]; __bf16 h[16]; };

static __device__ __forceinline__ v8f wmma_bf16(v16bf a, v16bf b, v8f c) {
  return __builtin_amdgcn_wmma_f32_16x16x32_bf16(
      /*neg_a=*/false, a, /*neg_b=*/false, b,
      /*c_mod=*/(short)0, c, /*reuse_a=*/false, /*reuse_b=*/false);
}

// A fragment (16x32 bf16) from LDS, row-major, base = &tile[row0][k0].
// This code's contiguous k order (k = 16*half + s): lane (l&15)=M; l<16 holds K=0..15, l>=16 holds K=16..31.
static __device__ __forceinline__ v16bf frag_a_lds(const __bf16* base, int ld) {
  const int lane = threadIdx.x & 31;
  const int m    = lane & 15;
  const int half = lane >> 4;
  const __bf16* p = base + m * ld + half * 16;
  FragBF u;
#pragma unroll
  for (int i = 0; i < 8; ++i) u.e[i] = p[i];
#pragma unroll
  for (int i = 0; i < 8; ++i) u.e[8 + i] = p[8 + i];
  return u.v;
}

// B fragment (32x16 bf16) where B[k][n] = tile[n][k] (n-major LDS tile).
// Contiguous k order (k = 16*half + s), as in frag_a_lds: lanes 0-15: N=lane, K=0..15; lanes 16-31: N=lane-16, K=16..31.
static __device__ __forceinline__ v16bf frag_b_nmajor(const __bf16* base, int ld) {
  const int lane = threadIdx.x & 31;
  const int n    = lane & 15;
  const int half = lane >> 4;
  const __bf16* p = base + n * ld + half * 16;
  FragBF u;
#pragma unroll
  for (int i = 0; i < 16; ++i) u.e[i] = p[i];
  return u.v;
}

static __device__ __forceinline__ void store_bf16x4(__bf16* dst, float4 f) {
  __bf16 h[4] = {(__bf16)f.x, (__bf16)f.y, (__bf16)f.z, (__bf16)f.w};
  *reinterpret_cast<uint2*>(dst) = *reinterpret_cast<const uint2*>(h);
}

#define LDA 72   // padded leading dim (bf16 elems): 144B rows, 16B aligned
#define LDB 72

// ---------------------------------------------------------------------------
// Kernel 1: Y_bf16[M,N] = bf16(X_f32[M,K]) * bf16(W_f32[K,N]) + bias
// Block tile 128x64, BK=64, 256 threads (8 waves), each wave 32x32.
// Software-pipelined: next k-tile loaded to registers behind current WMMAs.
// ---------------------------------------------------------------------------
__global__ __launch_bounds__(256) void proj_gemm(
    const float* __restrict__ X, const float* __restrict__ W,
    const float* __restrict__ bias, __bf16* __restrict__ Y,
    int M, int N, int K) {
  __shared__ __bf16 xs[128 * LDA];   // A tile, row-major (m x k)
  __shared__ __bf16 bt[64 * LDB];    // B tile, n-major  (n x k)

  const int t    = threadIdx.x;
  const int lane = t & 31;
  const int w    = t >> 5;
  const int wr   = w >> 1;           // 0..3 -> M offset 32*wr
  const int wc   = w & 1;            // 0..1 -> N offset 32*wc
  const int half = lane >> 4;
  const int nl   = lane & 15;
  const int m0   = blockIdx.x * 128;
  const int n0   = blockIdx.y * 64;

  // Staging assignments (fixed per thread).
  const int rowA = t >> 1, colA = (t & 1) * 32;      // X: 32 elems/thread
  const int rB   = t >> 2, cB0  = (t & 3) * 16;      // W: 16 elems/thread
  const float* Xrow = X + (size_t)(m0 + rowA) * K + colA;
  const float* Wcol = W + n0 + cB0;

  float4 xreg[8];
  float4 wreg[4];

  auto load_tiles = [&](int kk) {
    const float4* xp = reinterpret_cast<const float4*>(Xrow + kk);
#pragma unroll
    for (int v = 0; v < 8; ++v) xreg[v] = xp[v];
    const float4* wp =
        reinterpret_cast<const float4*>(Wcol + (size_t)(kk + rB) * N);
#pragma unroll
    for (int v = 0; v < 4; ++v) wreg[v] = wp[v];
  };

  auto stage_tiles = [&]() {
    __bf16* d = &xs[rowA * LDA + colA];
#pragma unroll
    for (int v = 0; v < 8; ++v) store_bf16x4(d + v * 4, xreg[v]);
#pragma unroll
    for (int v = 0; v < 4; ++v) {
      float4 f = wreg[v];
      bt[(cB0 + v * 4 + 0) * LDB + rB] = (__bf16)f.x;
      bt[(cB0 + v * 4 + 1) * LDB + rB] = (__bf16)f.y;
      bt[(cB0 + v * 4 + 2) * LDB + rB] = (__bf16)f.z;
      bt[(cB0 + v * 4 + 3) * LDB + rB] = (__bf16)f.w;
    }
  };

  v8f c00 = {}, c01 = {}, c10 = {}, c11 = {};

  load_tiles(0);
  for (int kk = 0; kk < K; kk += 64) {
    stage_tiles();
    __syncthreads();

    if (kk + 64 < K) load_tiles(kk + 64);                 // pipeline next tile
    if (kk + 192 < K) __builtin_prefetch(Xrow + kk + 192, 0, 0);  // HBM stream

#pragma unroll
    for (int ks = 0; ks < 64; ks += 32) {
      v16bf a0 = frag_a_lds(xs + (wr * 32 + 0)  * LDA + ks, LDA);
      v16bf a1 = frag_a_lds(xs + (wr * 32 + 16) * LDA + ks, LDA);
      v16bf b0 = frag_b_nmajor(bt + (wc * 32 + 0)  * LDB + ks, LDB);
      v16bf b1 = frag_b_nmajor(bt + (wc * 32 + 16) * LDB + ks, LDB);
      c00 = wmma_bf16(a0, b0, c00);
      c01 = wmma_bf16(a0, b1, c01);
      c10 = wmma_bf16(a1, b0, c10);
      c11 = wmma_bf16(a1, b1, c11);
    }
    __syncthreads();
  }

  // Bias add + store bf16. C layout: row = r + 8*half, col = nl.
  const int mbase = m0 + wr * 32;
  const int nbase = n0 + wc * 32;
  const float bn0 = bias[nbase + nl];
  const float bn1 = bias[nbase + 16 + nl];
  __bf16* yp  = Y + (size_t)(mbase + half * 8) * N + nbase + nl;
  __bf16* yp2 = yp + (size_t)16 * N;
#pragma unroll
  for (int r = 0; r < 8; ++r) {
    yp[0]   = (__bf16)(c00[r] + bn0);
    yp[16]  = (__bf16)(c01[r] + bn1);
    yp2[0]  = (__bf16)(c10[r] + bn0);
    yp2[16] = (__bf16)(c11[r] + bn1);
    yp += N; yp2 += N;
  }
}

// ---------------------------------------------------------------------------
// Kernel 2: streaming softmax. Block = (batch, 64 query rows), 128 thr / 4
// waves; wave w owns query rows 16w..16w+15. Pass 0: S = QK^T/sqrt(D) via
// WMMA, online row max/sum. Pass 1: recompute S, write P bf16 to scratch.
// Key tiles are row(key)-major, which IS the n-major layout for B = K^T.
// ---------------------------------------------------------------------------
__global__ __launch_bounds__(128) void attn_softmax(
    const __bf16* __restrict__ Q, const __bf16* __restrict__ Kmat,
    __bf16* __restrict__ P, int S, int D, float scale) {
  __shared__ __bf16 qs[64 * LDA];
  __shared__ __bf16 kt_[64 * LDA];

  const int t    = threadIdx.x;
  const int lane = t & 31;
  const int w    = t >> 5;
  const int half = lane >> 4;
  const int nl   = lane & 15;
  const int b    = blockIdx.y;
  const int q0   = blockIdx.x * 64;

  const __bf16* Qb = Q    + (size_t)b * S * D;
  const __bf16* Kb = Kmat + (size_t)b * S * D;
  __bf16*       Pb = P    + (size_t)b * S * S;

  const int row = t >> 1;
  const int col = (t & 1) * 32;
  const __bf16* Qrow = Qb + (size_t)(q0 + row) * D + col;

  uint4 qreg[4], kreg[4];

  float mrow[8], lrow[8];
#pragma unroll
  for (int r = 0; r < 8; ++r) { mrow[r] = -3.0e38f; lrow[r] = 0.0f; }

  for (int pass = 0; pass < 2; ++pass) {
    for (int kt0 = 0; kt0 < S; kt0 += 64) {
      const __bf16* Krow = Kb + (size_t)(kt0 + row) * D + col;

      auto load_tiles = [&](int d0) {
        const uint4* qp = reinterpret_cast<const uint4*>(Qrow + d0);
        const uint4* kp = reinterpret_cast<const uint4*>(Krow + d0);
#pragma unroll
        for (int v = 0; v < 4; ++v) { qreg[v] = qp[v]; kreg[v] = kp[v]; }
      };

      v8f c[4] = {{}, {}, {}, {}};

      load_tiles(0);
      for (int d0 = 0; d0 < D; d0 += 64) {
        {
          uint4* qd = reinterpret_cast<uint4*>(&qs[row * LDA + col]);
          uint4* kd = reinterpret_cast<uint4*>(&kt_[row * LDA + col]);
#pragma unroll
          for (int v = 0; v < 4; ++v) { qd[v] = qreg[v]; kd[v] = kreg[v]; }
        }
        __syncthreads();

        if (d0 + 64 < D) load_tiles(d0 + 64);             // pipeline next tile

#pragma unroll
        for (int ks = 0; ks < 64; ks += 32) {
          v16bf a = frag_a_lds(qs + (w * 16) * LDA + ks, LDA);
#pragma unroll
          for (int j = 0; j < 4; ++j) {
            v16bf bf = frag_b_nmajor(kt_ + (j * 16) * LDA + ks, LDA);
            c[j] = wmma_bf16(a, bf, c[j]);
          }
        }
        __syncthreads();
      }

#pragma unroll
      for (int j = 0; j < 4; ++j)
#pragma unroll
        for (int r = 0; r < 8; ++r) c[j][r] *= scale;

      if (pass == 0) {
        // Online row max / exp-sum. Row r+8*half spans 4 frags x 16 lanes.
#pragma unroll
        for (int r = 0; r < 8; ++r) {
          float mx = fmaxf(fmaxf(c[0][r], c[1][r]), fmaxf(c[2][r], c[3][r]));
#pragma unroll
          for (int off = 1; off < 16; off <<= 1)
            mx = fmaxf(mx, __shfl_xor(mx, off, 16));
          const float mnew = fmaxf(mrow[r], mx);
          float se = __expf(c[0][r] - mnew) + __expf(c[1][r] - mnew) +
                     __expf(c[2][r] - mnew) + __expf(c[3][r] - mnew);
#pragma unroll
          for (int off = 1; off < 16; off <<= 1)
            se += __shfl_xor(se, off, 16);
          lrow[r] = lrow[r] * __expf(mrow[r] - mnew) + se;
          mrow[r] = mnew;
        }
      } else {
        // Normalize and write P (bf16).
        __bf16* pp = Pb + (size_t)(q0 + w * 16 + half * 8) * S + kt0 + nl;
#pragma unroll
        for (int r = 0; r < 8; ++r) {
          const float inv = 1.0f / lrow[r];
#pragma unroll
          for (int j = 0; j < 4; ++j)
            pp[j * 16] = (__bf16)(__expf(c[j][r] - mrow[r]) * inv);
          pp += S;
        }
      }
    }
  }
}

// ---------------------------------------------------------------------------
// Kernel 3: O_f32[S,D] = P_bf16[S,S] * V_bf16[S,D] per batch.
// Same tiling/pipelining as proj_gemm; V tile transposed to n-major in LDS.
// ---------------------------------------------------------------------------
__global__ __launch_bounds__(256) void pv_gemm(
    const __bf16* __restrict__ P, const __bf16* __restrict__ V,
    float* __restrict__ O, int S, int D) {
  __shared__ __bf16 ps[128 * LDA];
  __shared__ __bf16 vt[64 * LDB];

  const int t    = threadIdx.x;
  const int lane = t & 31;
  const int w    = t >> 5;
  const int wr   = w >> 1;
  const int wc   = w & 1;
  const int half = lane >> 4;
  const int nl   = lane & 15;
  const int b    = blockIdx.z;
  const int m0   = blockIdx.x * 128;
  const int n0   = blockIdx.y * 64;

  const __bf16* Pb = P + (size_t)b * S * S;
  const __bf16* Vb = V + (size_t)b * S * D;
  float*        Ob = O + (size_t)b * S * D;

  const int rowA = t >> 1, colA = (t & 1) * 32;
  const int rB   = t >> 2, cB0  = (t & 3) * 16;
  const __bf16* Prow = Pb + (size_t)(m0 + rowA) * S + colA;
  const __bf16* Vcol = Vb + n0 + cB0;

  uint4 preg[4];
  PackBF16 vreg;

  auto load_tiles = [&](int kk) {
    const uint4* pp = reinterpret_cast<const uint4*>(Prow + kk);
#pragma unroll
    for (int v = 0; v < 4; ++v) preg[v] = pp[v];
    const uint4* vp =
        reinterpret_cast<const uint4*>(Vcol + (size_t)(kk + rB) * D);
    vreg.u[0] = vp[0]; vreg.u[1] = vp[1];
  };

  v8f c00 = {}, c01 = {}, c10 = {}, c11 = {};

  load_tiles(0);
  for (int kk = 0; kk < S; kk += 64) {
    {
      uint4* pd = reinterpret_cast<uint4*>(&ps[rowA * LDA + colA]);
#pragma unroll
      for (int v = 0; v < 4; ++v) pd[v] = preg[v];
#pragma unroll
      for (int i = 0; i < 16; ++i) vt[(cB0 + i) * LDB + rB] = vreg.h[i];
    }
    __syncthreads();

    if (kk + 64 < S) load_tiles(kk + 64);                  // pipeline next tile
    if (kk + 192 < S) __builtin_prefetch(Prow + kk + 192, 0, 0);

#pragma unroll
    for (int ks = 0; ks < 64; ks += 32) {
      v16bf a0 = frag_a_lds(ps + (wr * 32 + 0)  * LDA + ks, LDA);
      v16bf a1 = frag_a_lds(ps + (wr * 32 + 16) * LDA + ks, LDA);
      v16bf b0 = frag_b_nmajor(vt + (wc * 32 + 0)  * LDB + ks, LDB);
      v16bf b1 = frag_b_nmajor(vt + (wc * 32 + 16) * LDB + ks, LDB);
      c00 = wmma_bf16(a0, b0, c00);
      c01 = wmma_bf16(a0, b1, c01);
      c10 = wmma_bf16(a1, b0, c10);
      c11 = wmma_bf16(a1, b1, c11);
    }
    __syncthreads();
  }

  const int mbase = m0 + wr * 32;
  const int nbase = n0 + wc * 32;
  float* op  = Ob + (size_t)(mbase + half * 8) * D + nbase + nl;
  float* op2 = op + (size_t)16 * D;
#pragma unroll
  for (int r = 0; r < 8; ++r) {
    op[0]   = c00[r];
    op[16]  = c01[r];
    op2[0]  = c10[r];
    op2[16] = c11[r];
    op += D; op2 += D;
  }
}

// ---------------------------------------------------------------------------
extern "C" void kernel_launch(void* const* d_in, const int* in_sizes, int n_in,
                              void* d_out, int out_size, void* d_ws, size_t ws_size,
                              hipStream_t stream) {
  (void)in_sizes; (void)n_in; (void)out_size; (void)ws_size;
  const int Bn = 4, S = 2048, D = 1024;

  const float* query = (const float*)d_in[0];
  const float* key   = (const float*)d_in[1];
  const float* value = (const float*)d_in[2];
  const float* wq    = (const float*)d_in[3];
  const float* bq    = (const float*)d_in[4];
  const float* wk    = (const float*)d_in[5];
  const float* bk    = (const float*)d_in[6];
  const float* wv    = (const float*)d_in[7];
  const float* bv    = (const float*)d_in[8];

  // Scratch layout (bf16): Q | K | V | P  -> 16+16+16+32 = 80 MB.
  const size_t nQKV = (size_t)Bn * S * D;
  __bf16* Qb = (__bf16*)d_ws;
  __bf16* Kb = Qb + nQKV;
  __bf16* Vb = Kb + nQKV;
  __bf16* Pb = Vb + nQKV;

  const int M = Bn * S;  // 8192 flattened rows for the projections

  dim3 gProj(M / 128, D / 64);
  proj_gemm<<<gProj, 256, 0, stream>>>(query, wq, bq, Qb, M, D, D);
  proj_gemm<<<gProj, 256, 0, stream>>>(key,   wk, bk, Kb, M, D, D);
  proj_gemm<<<gProj, 256, 0, stream>>>(value, wv, bv, Vb, M, D, D);

  dim3 gAttn(S / 64, Bn);
  attn_softmax<<<gAttn, 128, 0, stream>>>(Qb, Kb, Pb, S, D, 0.03125f /* 1/sqrt(1024) */);

  dim3 gPV(S / 128, D / 64, Bn);
  pv_gemm<<<gPV, 256, 0, stream>>>(Pb, Vb, (float*)d_out, S, D);
}
